// DFTLayer_3255585210408
// MI455X (gfx1250) — matrix operand order repaired and checked by reading, not compiled, not run on hardware
//
#include <hip/hip_runtime.h>

// ---------------------------------------------------------------------------
// DFT-as-GEMM for MI455X (gfx1250, wave32, WMMA).
//   real[b,f,k] = sum_s x[b,f,s] * wcos[k,s]
//   imag[b,f,k] = sum_s x[b,f,s] * wsin[k,s]
//   out complex64 interleaved: (real, -imag)
// f32 inputs split into bf16 hi/lo ("bf16x3"), fed to
// v_wmma_f32_16x16x32_bf16 (8x the MACs/instr of the native f32 WMMA).
// Staging K-step = 64 (two WMMA substeps per barrier pair) to halve
// synchronization overhead per FLOP. Next-iteration global prefetch.
// ---------------------------------------------------------------------------

typedef __attribute__((ext_vector_type(16))) __bf16 v16bf;
typedef __attribute__((ext_vector_type(8)))  float  v8f;

#define NMAT   1024
#define NBATCH 16
#define BM     128   // block rows (f)
#define BN     64    // block cols (k)
#define BK     64    // staging K-step (s); 2 x WMMA K=32 substeps

union Frag32 {            // one 16x32 (A) or 32x16 (B) bf16 fragment per lane
    uint4 q[2];
    v16bf v;
};

struct __align__(8) US4 { unsigned short a, b, c, d; };

// Split f32 into bf16 hi + bf16 lo (x ~= hi + lo).
__device__ __forceinline__ void splitf(float x, unsigned short& h, unsigned short& l) {
    unsigned u  = __float_as_uint(x);
    unsigned uh = (u + 0x8000u) & 0xffff0000u;   // round-to-nearest bf16
    h = (unsigned short)(uh >> 16);
    float r = x - __uint_as_float(uh);
    l = (unsigned short)((__float_as_uint(r) + 0x8000u) >> 16);
}

extern "C" __global__ void __launch_bounds__(256, 1)
dft1024_wmma_bf16x3(const float* __restrict__ x,
                    const float* __restrict__ wsin,
                    const float* __restrict__ wcos,
                    float* __restrict__ out)
{
    // X tile row-major [BM][BK]; W tiles stored [BN][BK] (k rows, s contiguous) so the
    // B fragment (lane = N column, elements = K) is two contiguous ds_load_b128.
    __shared__ alignas(16) unsigned short XsH[BM * BK];
    __shared__ alignas(16) unsigned short XsL[BM * BK];
    __shared__ alignas(16) unsigned short WcH[BK * BN];
    __shared__ alignas(16) unsigned short WcL[BK * BN];
    __shared__ alignas(16) unsigned short WsH[BK * BN];
    __shared__ alignas(16) unsigned short WsL[BK * BN];

    const int tid  = threadIdx.x;
    const int lane = tid & 31;
    const int wid  = tid >> 5;        // 0..7
    const int wm   = wid >> 1;        // 0..3 : 32-row strip within BM
    const int wn   = wid & 1;         // 0..1 : 32-col strip within BN

    const int nb = blockIdx.x;        // 16 tiles of BN cols (k)
    const int mb = blockIdx.y;        // 8  tiles of BM rows (f)
    const int b  = blockIdx.z;        // batch

    const float* xB    = x + ((size_t)b * NMAT + (size_t)mb * BM) * NMAT;
    const int    n0blk = nb * BN;

    const v8f vzero = {0.f, 0.f, 0.f, 0.f, 0.f, 0.f, 0.f, 0.f};
    v8f accC[2][2], accS[2][2];
#pragma unroll
    for (int i = 0; i < 2; ++i)
#pragma unroll
        for (int j = 0; j < 2; ++j) { accC[i][j] = vzero; accS[i][j] = vzero; }

    for (int ks = 0; ks < NMAT; ks += BK) {
        __syncthreads();

        // ---- prefetch next iteration's cachelines (global_prefetch_b8) ----
        if (ks + BK < NMAT) {
            const int ksn = ks + BK;
            {   // X: 128 rows x 2 lines (64 floats) -> 256 lines, 1/thread
                const int row = tid >> 1;
                const int off = (tid & 1) << 5;
                __builtin_prefetch(xB + (size_t)row * NMAT + ksn + off, 0, 1);
            }
            {   // W: 64 rows x 2 lines x 2 matrices -> 256 lines, 1/thread
                const int kr  = tid >> 2;
                const int sel = tid & 3;
                const int off = (sel & 1) << 5;
                const float* wp = (sel >> 1) ? wsin : wcos;
                __builtin_prefetch(wp + (size_t)(n0blk + kr) * NMAT + ksn + off, 0, 1);
            }
        }

        // ---- stage X tile: BM x BK f32 -> bf16 hi/lo (2048 float4, 8/thread)
#pragma unroll
        for (int i = 0; i < 8; ++i) {
            const int f   = tid + i * 256;     // 0..2047
            const int row = f >> 4;            // 0..127
            const int c4  = (f & 15) << 2;     // 0,4,..,60
            const float4 v = *(const float4*)(xB + (size_t)row * NMAT + ks + c4);
            unsigned short h0,l0,h1,l1,h2,l2,h3,l3;
            splitf(v.x,h0,l0); splitf(v.y,h1,l1); splitf(v.z,h2,l2); splitf(v.w,h3,l3);
            const int o = row * BK + c4;
            *(US4*)&XsH[o] = US4{h0, h1, h2, h3};
            *(US4*)&XsL[o] = US4{l0, l1, l2, l3};
        }

        // ---- stage W tiles (cos & sin) as [k][s] (s contiguous) (1024 float4 ea, 4/thread)
#pragma unroll
        for (int i = 0; i < 4; ++i) {
            const int f  = tid + i * 256;      // 0..1023
            const int kr = f >> 4;             // 0..63  (k within tile)
            const int s4 = (f & 15) << 2;      // 0,4,..,60 (s within tile)
            const size_t go = (size_t)(n0blk + kr) * NMAT + ks + s4;
            const float4 vc = *(const float4*)(wcos + go);
            const float4 vs = *(const float4*)(wsin + go);
            const float cf[4] = {vc.x, vc.y, vc.z, vc.w};
            const float sf[4] = {vs.x, vs.y, vs.z, vs.w};
#pragma unroll
            for (int j = 0; j < 4; ++j) {
                unsigned short h, l;
                const int o = kr * BK + s4 + j;
                splitf(cf[j], h, l); WcH[o] = h; WcL[o] = l;
                splitf(sf[j], h, l); WsH[o] = h; WsL[o] = l;
            }
        }
        __syncthreads();

        // ---- two WMMA K=32 substeps over the staged BK=64 slab ----
#pragma unroll
        for (int kk = 0; kk < BK; kk += 32) {
            // A fragments (16x32 bf16): lane = M (0..15), lane>=16 -> K+8 group
            const int kb   = (lane >> 4) << 3;     // 0 or 8
            const int mrow = lane & 15;
            Frag32 aH[2], aL[2];
#pragma unroll
            for (int tm = 0; tm < 2; ++tm) {
                const unsigned short* p = &XsH[(wm * 32 + tm * 16 + mrow) * BK + kk];
                aH[tm].q[0] = *(const uint4*)(p + kb);
                aH[tm].q[1] = *(const uint4*)(p + kb + 16);
                const unsigned short* q = &XsL[(wm * 32 + tm * 16 + mrow) * BK + kk];
                aL[tm].q[0] = *(const uint4*)(q + kb);
                aL[tm].q[1] = *(const uint4*)(q + kb + 16);
            }

            // B fragments + WMMA (bf16x3: hh + hl + lh)
#pragma unroll
            for (int tn = 0; tn < 2; ++tn) {
                const int nc = wn * 32 + tn * 16;
                Frag32 bcH, bcL, bsH, bsL;
                const unsigned short* pc = &WcH[(nc + mrow) * BK + kk];
                bcH.q[0] = *(const uint4*)(pc + kb);   bcH.q[1] = *(const uint4*)(pc + kb + 16);
                const unsigned short* ql = &WcL[(nc + mrow) * BK + kk];
                bcL.q[0] = *(const uint4*)(ql + kb);   bcL.q[1] = *(const uint4*)(ql + kb + 16);
                const unsigned short* ps = &WsH[(nc + mrow) * BK + kk];
                bsH.q[0] = *(const uint4*)(ps + kb);   bsH.q[1] = *(const uint4*)(ps + kb + 16);
                const unsigned short* qs = &WsL[(nc + mrow) * BK + kk];
                bsL.q[0] = *(const uint4*)(qs + kb);   bsL.q[1] = *(const uint4*)(qs + kb + 16);

#pragma unroll
                for (int tm = 0; tm < 2; ++tm) {
                    accC[tm][tn] = __builtin_amdgcn_wmma_f32_16x16x32_bf16(
                        false, aH[tm].v, false, bcH.v, (short)0, accC[tm][tn], false, false);
                    accC[tm][tn] = __builtin_amdgcn_wmma_f32_16x16x32_bf16(
                        false, aH[tm].v, false, bcL.v, (short)0, accC[tm][tn], false, false);
                    accC[tm][tn] = __builtin_amdgcn_wmma_f32_16x16x32_bf16(
                        false, aL[tm].v, false, bcH.v, (short)0, accC[tm][tn], false, false);

                    accS[tm][tn] = __builtin_amdgcn_wmma_f32_16x16x32_bf16(
                        false, aH[tm].v, false, bsH.v, (short)0, accS[tm][tn], false, false);
                    accS[tm][tn] = __builtin_amdgcn_wmma_f32_16x16x32_bf16(
                        false, aH[tm].v, false, bsL.v, (short)0, accS[tm][tn], false, false);
                    accS[tm][tn] = __builtin_amdgcn_wmma_f32_16x16x32_bf16(
                        false, aL[tm].v, false, bsH.v, (short)0, accS[tm][tn], false, false);
                }
            }
        }
    }

    // ---- epilogue: C/D layout VGPR r -> M = r (+8 for lanes 16..31), N = lane&15
    const int mhalf = (lane >> 4) << 3;   // 0 or 8
    const int ncl   = lane & 15;
#pragma unroll
    for (int tm = 0; tm < 2; ++tm) {
#pragma unroll
        for (int tn = 0; tn < 2; ++tn) {
            const int col = n0blk + wn * 32 + tn * 16 + ncl;
#pragma unroll
            for (int r = 0; r < 8; ++r) {
                const int row = mb * BM + wm * 32 + tm * 16 + mhalf + r;
                const size_t o = (((size_t)b * NMAT + row) * NMAT + col) * 2;
                float2 v;
                v.x =  accC[tm][tn][r];    // real
                v.y = -accS[tm][tn][r];    // -imag
                *(float2*)(out + o) = v;
            }
        }
    }
}

extern "C" void kernel_launch(void* const* d_in, const int* in_sizes, int n_in,
                              void* d_out, int out_size, void* d_ws, size_t ws_size,
                              hipStream_t stream) {
    (void)in_sizes; (void)n_in; (void)out_size; (void)d_ws; (void)ws_size;
    const float* x    = (const float*)d_in[0];   // [16,1024,1024]
    const float* wsin = (const float*)d_in[1];   // [1024,1024]
    const float* wcos = (const float*)d_in[2];   // [1024,1024]
    float* out = (float*)d_out;                  // complex64 interleaved

    dim3 grid(NMAT / BN, NMAT / BM, NBATCH);     // (16, 8, 16)
    dim3 block(256);                             // 8 wave32 waves
    hipLaunchKernelGGL(dft1024_wmma_bf16x3, grid, block, 0, stream,
                       x, wsin, wcos, out);
}
